// SelfAttentionLayer_27462020890950
// MI455X (gfx1250) — matrix operand order repaired and checked by reading, not compiled, not run on hardware
//
#include <hip/hip_runtime.h>

#define ATHENA_USE_TDM (__has_builtin(__builtin_amdgcn_tensor_load_to_lds))

#if __has_builtin(__builtin_amdgcn_sched_barrier)
#define SCHED_BARRIER() __builtin_amdgcn_sched_barrier(0)
#else
#define SCHED_BARRIER()
#endif

typedef __attribute__((ext_vector_type(16))) _Float16 v16h;
typedef __attribute__((ext_vector_type(8)))  _Float16 v8h;
typedef __attribute__((ext_vector_type(4)))  _Float16 v4h;
typedef __attribute__((ext_vector_type(8)))  float    v8f;
typedef __attribute__((ext_vector_type(4)))  float    v4f;
typedef __attribute__((ext_vector_type(4)))  unsigned int u32x4;
typedef __attribute__((ext_vector_type(4)))  int      i32x4;
typedef __attribute__((ext_vector_type(8)))  int      i32x8;

#define BDIM 32
#define CDIM 256
#define DDIM 256
#define NDIM 4096

// ---------------- WMMA helpers (CDNA5 16x16x32 f16 -> f32) ----------------

__device__ __forceinline__ v8f wmma32(const v16h a, const v16h b, const v8f c) {
  return __builtin_amdgcn_wmma_f32_16x16x32_f16(false, a, false, b, (short)0, c,
                                                false, false);
}

// A fragment 16x32 (MxK), row-major src, leading dim `ld` halves.
__device__ __forceinline__ v16h frag_a_g(const _Float16* src, int ld, int lane) {
  const int m  = lane & 15;
  const int kb = (lane & 16) ? 8 : 0;
  const _Float16* p = src + m * ld + kb;
  v8h lo = *(const v8h*)(p);
  v8h hi = *(const v8h*)(p + 16);
  v16h a;
#pragma unroll
  for (int i = 0; i < 8; ++i) { a[i] = lo[i]; a[i + 8] = hi[i]; }
  return a;
}

// B fragment 32x16 (KxN); element (n,k) at src[n*ld + k] (K contiguous).
__device__ __forceinline__ v16h frag_b_rows(const _Float16* src, int ld, int lane) {
  const int n  = lane & 15;
  const int kb = (lane & 16) ? 8 : 0;
  const _Float16* p = src + n * ld + kb;
  v8h lo = *(const v8h*)(p);
  v8h hi = *(const v8h*)(p + 16);
  v16h b;
#pragma unroll
  for (int i = 0; i < 8; ++i) { b[i] = lo[i]; b[i + 8] = hi[i]; }
  return b;
}

// 16x16 f32 C-fragment -> row-major store via per-wave LDS transpose.
__device__ __forceinline__ void tile_store_f32(float* dst, int ld, float* stg,
                                               const v8f c, int lane) {
  const int n = lane & 15;
  const int moff = (lane & 16) ? 8 : 0;
#pragma unroll
  for (int r = 0; r < 8; ++r) stg[(moff + r) * 20 + n] = c[r];
  __builtin_amdgcn_wave_barrier();   // same-wave DS ops are in-order
  const int r2 = lane >> 1;
  const int h8 = (lane & 1) * 8;
  const float* sp = stg + r2 * 20 + h8;
  v4f lo = *(const v4f*)sp;
  v4f hi = *(const v4f*)(sp + 4);
  v8f row;
#pragma unroll
  for (int i = 0; i < 4; ++i) { row[i] = lo[i]; row[4 + i] = hi[i]; }
  *(v8f*)(dst + r2 * ld + h8) = row;
  __builtin_amdgcn_wave_barrier();
}

__device__ __forceinline__ void tile_store_f16(_Float16* dst, int ld, float* stg,
                                               const v8f c, int lane) {
  const int n = lane & 15;
  const int moff = (lane & 16) ? 8 : 0;
#pragma unroll
  for (int r = 0; r < 8; ++r) stg[(moff + r) * 20 + n] = c[r];
  __builtin_amdgcn_wave_barrier();
  const int r2 = lane >> 1;
  const int h8 = (lane & 1) * 8;
  const float* sp = stg + r2 * 20 + h8;
  v4f lo = *(const v4f*)sp;
  v4f hi = *(const v4f*)(sp + 4);
  v8h row;
#pragma unroll
  for (int i = 0; i < 4; ++i) {
    row[i]     = (_Float16)lo[i];
    row[4 + i] = (_Float16)hi[i];
  }
  *(v8h*)(dst + r2 * ld + h8) = row;
  __builtin_amdgcn_wave_barrier();
}

// ---------------- TDM: 2-D tile (128 rows x 16 dwords) global -> LDS ---------
// pad_enable, interval = 16 dwords, amount = 4 dwords  => LDS row stride = 20
// dwords = 40 halves, matching frag_b_rows(ld=40).

struct TdmDesc {
  u32x4 g0;
  i32x8 g1;
};

__device__ __forceinline__ TdmDesc tdm_make_desc(unsigned row_stride_dw) {
  TdmDesc d;
  d.g0.x = 1u;                                  // count=1, user D#
  d.g0.y = 0u;                                  // lds_addr (patched per issue)
  d.g0.z = 0u;                                  // global_addr lo (patched)
  d.g0.w = (2u << 30);                          // type=2 (patched | addr hi)
  const unsigned td0 = row_stride_dw;           // tensor_dim0 (dwords)
  const unsigned td1 = (1u << 20);              // tensor_dim1 (large, no OOB)
  d.g1[0] = (int)((2u << 16)        // data_size = 4B
                | (1u << 20)        // pad_enable
                | (3u << 22)        // pad_interval: 16 dwords
                | (3u << 25));      // pad_amount:   4 dwords
  d.g1[1] = (int)((td0 & 0xFFFFu) << 16);
  d.g1[2] = (int)(((td0 >> 16) & 0xFFFFu) | ((td1 & 0xFFFFu) << 16));
  d.g1[3] = (int)(((td1 >> 16) & 0xFFFFu) | (16u << 16));  // tile_dim0 = 16 dw
  d.g1[4] = 128;                                           // tile_dim1 = 128 rows
  d.g1[5] = (int)row_stride_dw;                            // dim0 stride lo32
  d.g1[6] = 0;
  d.g1[7] = 0;
  return d;
}

__device__ __forceinline__ void tdm_issue(const TdmDesc& d, unsigned lds_off,
                                          const void* gptr) {
#if ATHENA_USE_TDM
  const unsigned long long ga = (unsigned long long)(size_t)gptr;
  u32x4 g0 = d.g0;
  g0.y = lds_off;
  g0.z = (unsigned)ga;
  g0.w = (unsigned)((ga >> 32) & 0x01FFFFFFull) | (2u << 30);
  i32x4 z4 = {0, 0, 0, 0};
#if __clang_major__ >= 23
  i32x8 z8 = {0, 0, 0, 0, 0, 0, 0, 0};
  __builtin_amdgcn_tensor_load_to_lds(g0, d.g1, z4, z4, z8, 0);
#else
  __builtin_amdgcn_tensor_load_to_lds(g0, d.g1, z4, z4, 0);
#endif
#else
  (void)d; (void)lds_off; (void)gptr;
#endif
}

// ---------------- Kernel 0: convert weights to f16 [3][D][C] ----------------

__global__ void wcvt_kernel(const float* __restrict__ Wq, const float* __restrict__ Wk,
                            const float* __restrict__ Wv, _Float16* __restrict__ Wf) {
  const int i = blockIdx.x * 256 + threadIdx.x;
  Wf[i]             = (_Float16)Wq[i];
  Wf[65536 + i]     = (_Float16)Wk[i];
  Wf[2 * 65536 + i] = (_Float16)Wv[i];
}

// ---------------- Kernel 1: fused QKV projection -----------------------------

__launch_bounds__(256)
__global__ void proj_kernel(const float* __restrict__ x,
                            const _Float16* __restrict__ Wf,
                            const float* __restrict__ bq,
                            const float* __restrict__ bk,
                            const float* __restrict__ bv,
                            _Float16* __restrict__ Q,
                            _Float16* __restrict__ Kc,
                            _Float16* __restrict__ Vt) {
  __shared__ __align__(32) _Float16 xT[64 * 264];   // [n][c], pad 8 halves
  __shared__ __align__(16) float stg[8 * 16 * 20];
  const int b    = blockIdx.y;
  const int n0   = blockIdx.x * 64;
  const int t    = threadIdx.x;
  const int wave = t >> 5;
  const int lane = t & 31;

  {  // x tile [256 c][64 n] f32 -> xT[n][c] f16; 4 c-rows x 16 n per thread,
     // packed ds_store_b64 (conflict-free: lane l -> dword banks 2l,2l+1).
    const int c0 = (t & 63) * 4;
    const int nh = (t >> 6) * 16;
    const float* xp = x + ((size_t)b * CDIM + c0) * NDIM + n0 + nh;
#pragma unroll
    for (int j = 0; j < 4; ++j) {
      v4f r0 = *(const v4f*)(xp + 0 * NDIM + j * 4);
      v4f r1 = *(const v4f*)(xp + 1 * NDIM + j * 4);
      v4f r2 = *(const v4f*)(xp + 2 * NDIM + j * 4);
      v4f r3 = *(const v4f*)(xp + 3 * NDIM + j * 4);
#pragma unroll
      for (int u = 0; u < 4; ++u) {
        v4h p;
        p[0] = (_Float16)r0[u];
        p[1] = (_Float16)r1[u];
        p[2] = (_Float16)r2[u];
        p[3] = (_Float16)r3[u];
        *(v4h*)(&xT[(nh + j * 4 + u) * 264 + c0]) = p;
      }
    }
  }
  __syncthreads();

  float* mystg = stg + wave * (16 * 20);
  for (int task = wave; task < 48; task += 8) {   // 6 uniform tasks per wave
    const int mat = task >> 4;                    // 0=Q 1=K 2=V
    const int dt  = task & 15;
    const float* bias = (mat == 0) ? bq : ((mat == 1) ? bk : bv);
    const int moff = (lane & 16) ? 8 : 0;
    v8f binit;
#pragma unroll
    for (int r = 0; r < 8; ++r) binit[r] = bias[dt * 16 + moff + r];
    v8f acc[4];
#pragma unroll
    for (int nt = 0; nt < 4; ++nt) acc[nt] = binit;

    const _Float16* wsrc = Wf + mat * (DDIM * CDIM) + dt * 16 * CDIM;
#pragma unroll
    for (int kk = 0; kk < 8; ++kk) {
      v16h a = frag_a_g(wsrc + kk * 32, CDIM, lane);
      v16h bb[4];
#pragma unroll
      for (int nt = 0; nt < 4; ++nt)
        bb[nt] = frag_b_rows(xT + nt * 16 * 264 + kk * 32, 264, lane);
      SCHED_BARRIER();                            // issue all loads before WMMAs
#pragma unroll
      for (int nt = 0; nt < 4; ++nt) acc[nt] = wmma32(a, bb[nt], acc[nt]);
    }

    if (mat == 2) {                               // V -> [B][N][D] (free transpose)
      const int n = lane & 15;
#pragma unroll
      for (int nt = 0; nt < 4; ++nt) {
        v8h h;
#pragma unroll
        for (int r = 0; r < 8; ++r) h[r] = (_Float16)acc[nt][r];
        *(v8h*)(Vt + ((size_t)b * NDIM + n0 + nt * 16 + n) * DDIM + dt * 16 + moff) = h;
      }
    } else {                                      // Q/K -> [B][D][N]
      _Float16* dst = ((mat == 0) ? Q : Kc) + ((size_t)b * DDIM + dt * 16) * NDIM + n0;
#pragma unroll
      for (int nt = 0; nt < 4; ++nt)
        tile_store_f16(dst + nt * 16, NDIM, mystg, acc[nt], lane);
    }
  }
}

// ---------------- Kernel 2: scores = Q K^T (K-loop over N=4096) --------------

__launch_bounds__(256)
__global__ void scores_kernel(const _Float16* __restrict__ Q,
                              const _Float16* __restrict__ Kc,
                              float* __restrict__ S) {
  __shared__ __align__(32) _Float16 Kl[2][128 * 40];  // [e][32 n] rows, stride 40
  __shared__ __align__(16) float stg[8 * 16 * 20];
  const int b     = blockIdx.y;
  const int dbase = (blockIdx.x & 1) * 128;
  const int ebase = (blockIdx.x >> 1) * 128;
  const int t = threadIdx.x, wave = t >> 5, lane = t & 31;
  const _Float16* qsrc = Q + ((size_t)b * DDIM + dbase + wave * 16) * NDIM;
  const _Float16* kbg  = Kc + ((size_t)b * DDIM + ebase) * NDIM;

  v8f acc[8];
#pragma unroll
  for (int e = 0; e < 8; ++e)
#pragma unroll
    for (int r = 0; r < 8; ++r) acc[e][r] = 0.f;

#if ATHENA_USE_TDM
  const TdmDesc kd = tdm_make_desc(2048);   // K row stride: 4096 halves
  const unsigned ldsK0 = (unsigned)(size_t)&Kl[0][0];
  const unsigned ldsK1 = (unsigned)(size_t)&Kl[1][0];
  if (wave == 0) tdm_issue(kd, ldsK0, kbg);
  for (int kk = 0; kk < 128; kk += 2) {
#pragma unroll
    for (int half = 0; half < 2; ++half) {      // buffer index is literal
      const int k = kk + half;
      if (wave == 0) __builtin_amdgcn_s_wait_tensorcnt(0);
      __syncthreads();
      if (wave == 0 && k + 1 < 128)
        tdm_issue(kd, half ? ldsK0 : ldsK1, kbg + (size_t)(k + 1) * 32);
      const _Float16* kl = half ? &Kl[1][0] : &Kl[0][0];
      v16h a = frag_a_g(qsrc + (size_t)k * 32, NDIM, lane);
      if (k + 8 < 128)
        __builtin_prefetch(qsrc + (size_t)(k + 8) * 32 + (size_t)(lane & 15) * NDIM, 0, 0);
      v16h bb[8];
#pragma unroll
      for (int et = 0; et < 8; ++et) bb[et] = frag_b_rows(kl + et * 16 * 40, 40, lane);
      SCHED_BARRIER();
#pragma unroll
      for (int et = 0; et < 8; ++et) acc[et] = wmma32(a, bb[et], acc[et]);
    }
  }
#else
  const int le = t >> 1, lh = (t & 1) * 16;
  const _Float16* ksrc = kbg + (size_t)le * NDIM + lh;
  v16h pre = *(const v16h*)(ksrc);
  int buf = 0;
  for (int kk = 0; kk < 128; ++kk) {
    *(v16h*)(&Kl[buf][le * 40 + lh]) = pre;
    if (kk + 1 < 128) pre = *(const v16h*)(ksrc + (size_t)(kk + 1) * 32);
    __syncthreads();
    v16h a = frag_a_g(qsrc + (size_t)kk * 32, NDIM, lane);
    v16h bb[8];
#pragma unroll
    for (int et = 0; et < 8; ++et) bb[et] = frag_b_rows(&Kl[buf][et * 16 * 40], 40, lane);
    SCHED_BARRIER();
#pragma unroll
    for (int et = 0; et < 8; ++et) acc[et] = wmma32(a, bb[et], acc[et]);
    buf ^= 1;
  }
#endif

  float* sdst = S + ((size_t)b * DDIM + dbase + wave * 16) * DDIM + ebase;
  float* mystg = stg + wave * (16 * 20);
#pragma unroll
  for (int et = 0; et < 8; ++et)
    tile_store_f32(sdst + et * 16, DDIM, mystg, acc[et], lane);
}

// ---------------- Kernel 3: softmax rows (scale 1/64), emit f16 weights ------

__launch_bounds__(256)
__global__ void softmax_kernel(const float* __restrict__ S, _Float16* __restrict__ Wm) {
  const int row  = blockIdx.x * 8 + (threadIdx.x >> 5);  // wave per row
  const int lane = threadIdx.x & 31;
  v8f v = *(const v8f*)(S + (size_t)row * DDIM + lane * 8);
  const float scale = 0.015625f;  // 1/sqrt(4096)
  float m = -3.4e38f;
#pragma unroll
  for (int i = 0; i < 8; ++i) { v[i] *= scale; m = fmaxf(m, v[i]); }
#pragma unroll
  for (int o = 16; o > 0; o >>= 1) m = fmaxf(m, __shfl_xor(m, o, 32));
  float s = 0.f;
#pragma unroll
  for (int i = 0; i < 8; ++i) { v[i] = __expf(v[i] - m); s += v[i]; }
#pragma unroll
  for (int o = 16; o > 0; o >>= 1) s += __shfl_xor(s, o, 32);
  const float inv = 1.f / s;
  v8h h;
#pragma unroll
  for (int i = 0; i < 8; ++i) h[i] = (_Float16)(v[i] * inv);
  *(v8h*)(Wm + (size_t)row * DDIM + lane * 8) = h;
}

// ---------------- Kernel 4: out = weights @ V (K = 256) ----------------------

__launch_bounds__(256)
__global__ void out_kernel(const _Float16* __restrict__ Wm,
                           const _Float16* __restrict__ Vt,
                           float* __restrict__ out) {
  __shared__ __align__(32) _Float16 Vl[2][128 * 40];  // [n][32 e] rows, stride 40
  __shared__ __align__(16) float stg[8 * 16 * 20];
  const int b     = blockIdx.z;
  const int nbase = blockIdx.x * 128;
  const int dbase = blockIdx.y * 128;
  const int t = threadIdx.x, wave = t >> 5, lane = t & 31;
  const _Float16* vbg  = Vt + ((size_t)b * NDIM + nbase) * DDIM;
  const _Float16* asrc = Wm + ((size_t)b * DDIM + dbase + wave * 16) * DDIM;

  v8f acc[8];
#pragma unroll
  for (int e = 0; e < 8; ++e)
#pragma unroll
    for (int r = 0; r < 8; ++r) acc[e][r] = 0.f;

#if ATHENA_USE_TDM
  const TdmDesc vd = tdm_make_desc(128);    // Vt row stride: 256 halves
  const unsigned ldsV0 = (unsigned)(size_t)&Vl[0][0];
  const unsigned ldsV1 = (unsigned)(size_t)&Vl[1][0];
  if (wave == 0) tdm_issue(vd, ldsV0, vbg);
  for (int kk = 0; kk < 8; kk += 2) {
#pragma unroll
    for (int half = 0; half < 2; ++half) {
      const int k = kk + half;
      if (wave == 0) __builtin_amdgcn_s_wait_tensorcnt(0);
      __syncthreads();
      if (wave == 0 && k + 1 < 8)
        tdm_issue(vd, half ? ldsV0 : ldsV1, vbg + (size_t)(k + 1) * 32);
      const _Float16* vl = half ? &Vl[1][0] : &Vl[0][0];
      v16h a = frag_a_g(asrc + k * 32, DDIM, lane);
      v16h bb[8];
#pragma unroll
      for (int nt = 0; nt < 8; ++nt) bb[nt] = frag_b_rows(vl + nt * 16 * 40, 40, lane);
      SCHED_BARRIER();
#pragma unroll
      for (int nt = 0; nt < 8; ++nt) acc[nt] = wmma32(a, bb[nt], acc[nt]);
    }
  }
#else
  const int ln = t >> 1, lh = (t & 1) * 16;
  const _Float16* vsrc = vbg + (size_t)ln * DDIM + lh;
  v16h pre = *(const v16h*)(vsrc);
  int buf = 0;
#pragma unroll
  for (int kk = 0; kk < 8; ++kk) {
    *(v16h*)(&Vl[buf][ln * 40 + lh]) = pre;
    if (kk + 1 < 8) pre = *(const v16h*)(vsrc + (size_t)(kk + 1) * 32);
    __syncthreads();
    v16h a = frag_a_g(asrc + kk * 32, DDIM, lane);
    v16h bb[8];
#pragma unroll
    for (int nt = 0; nt < 8; ++nt) bb[nt] = frag_b_rows(&Vl[buf][nt * 16 * 40], 40, lane);
    SCHED_BARRIER();
#pragma unroll
    for (int nt = 0; nt < 8; ++nt) acc[nt] = wmma32(a, bb[nt], acc[nt]);
    buf ^= 1;
  }
#endif

  float* dst = out + ((size_t)b * DDIM + dbase + wave * 16) * NDIM + nbase;
  float* mystg = stg + wave * (16 * 20);
#pragma unroll
  for (int nt = 0; nt < 8; ++nt)
    tile_store_f32(dst + nt * 16, NDIM, mystg, acc[nt], lane);
}

// ---------------- Launch ------------------------------------------------------

extern "C" void kernel_launch(void* const* d_in, const int* in_sizes, int n_in,
                              void* d_out, int out_size, void* d_ws, size_t ws_size,
                              hipStream_t stream) {
  const float* x  = (const float*)d_in[0];
  const float* Wq = (const float*)d_in[1];
  const float* bq = (const float*)d_in[2];
  const float* Wk = (const float*)d_in[3];
  const float* bk = (const float*)d_in[4];
  const float* Wv = (const float*)d_in[5];
  const float* bv = (const float*)d_in[6];
  float* out = (float*)d_out;

  char* ws = (char*)d_ws;
  const size_t szW   = (size_t)3 * DDIM * CDIM * 2;
  const size_t szQKV = (size_t)BDIM * DDIM * NDIM * 2;   // 64 MiB each
  const size_t szS   = (size_t)BDIM * DDIM * DDIM * 4;   // 8 MiB
  _Float16* Wf = (_Float16*)(ws);
  _Float16* Q  = (_Float16*)(ws + szW);
  _Float16* Kc = (_Float16*)(ws + szW + szQKV);
  _Float16* Vt = (_Float16*)(ws + szW + 2 * szQKV);
  float*    S  = (float*)   (ws + szW + 3 * szQKV);
  _Float16* Wm = (_Float16*)(ws + szW + 3 * szQKV + szS);

  wcvt_kernel   <<<dim3(256),       dim3(256), 0, stream>>>(Wq, Wk, Wv, Wf);
  proj_kernel   <<<dim3(64, 32),    dim3(256), 0, stream>>>(x, Wf, bq, bk, bv, Q, Kc, Vt);
  scores_kernel <<<dim3(4, 32),     dim3(256), 0, stream>>>(Q, Kc, S);
  softmax_kernel<<<dim3(1024),      dim3(256), 0, stream>>>(S, Wm);
  out_kernel    <<<dim3(32, 2, 32), dim3(256), 0, stream>>>(Wm, Vt, out);
}
